// ODELSTMCell_9947144257770
// MI455X (gfx1250) — matrix operand order repaired and checked by reading, not compiled, not run on hardware
//
#include <hip/hip_runtime.h>

// ---------------------------------------------------------------------------
// ODE-LSTM fused kernel for MI455X (gfx1250, wave32, WMMA bf16 16x16x32)
//
// z = x@Wih^T + h@Whh^T + b  -> gates -> (h,c), twice; then RK4 with
// f(h) = tanh(h@W1^T + b1)@W2^T + b2.  All GEMMs are act @ W^T with W in
// torch [out,in] row-major layout, so both WMMA operands load contiguous
// 16B chunks (A from LDS activations, B straight from L2-resident bf16
// weights) -- no transposes needed.  Nonlinearities use the gfx1250
// hardware V_TANH_F32 transcendental (sigmoid via 0.5+0.5*tanh(x/2)).
// ---------------------------------------------------------------------------

typedef __bf16 bf16_t;
typedef __attribute__((ext_vector_type(16))) __bf16 bf16x16;
typedef __attribute__((ext_vector_type(8)))  float  f32x8;

#define B_TOT 32768
#define I_DIM 256
#define H_DIM 512
#define MT    32        // batch rows per block
#define XPAD  264       // 256 + 8 bf16 pad (row stride 528B -> 16B aligned, bank-skewed)
#define HPAD  520       // 512 + 8 bf16 pad (row stride 1040B -> 16B aligned)

__device__ __forceinline__ f32x8 wmma_bf16(bf16x16 a, bf16x16 b, f32x8 c) {
  return __builtin_amdgcn_wmma_f32_16x16x32_bf16(
      /*neg_a=*/false, a, /*neg_b=*/false, b,
      /*c_mod=*/(short)0, c, /*reuse_a=*/false, /*reuse_b=*/false);
}

// Hardware tanh (gfx1250 V_TANH_F32); 1 TRANS op instead of ~18 VALU via OCML.
__device__ __forceinline__ float fast_tanh(float x) {
#if __has_builtin(__builtin_amdgcn_tanhf)
  return __builtin_amdgcn_tanhf(x);
#elif __has_builtin(__builtin_amdgcn_tanh_f32)
  return __builtin_amdgcn_tanh_f32(x);
#else
  return tanhf(x);
#endif
}

// sigmoid(x) == 0.5 + 0.5*tanh(0.5*x)  (exact identity)
__device__ __forceinline__ float sigf(float x) {
  return __builtin_fmaf(0.5f, fast_tanh(0.5f * x), 0.5f);
}

// A-fragment (16x32 bf16): lane l holds row (l&15); lanes<16: K=[kk..kk+15],
// lanes>=16: K=[kk+16..kk+31] (16 consecutive).  Two 16B loads.
__device__ __forceinline__ bf16x16 fragA(const bf16_t* rowp, int kk, int hi8) {
  union { uint4 u[2]; bf16x16 v; } c;
  c.u[0] = *(const uint4*)(rowp + kk + hi8 * 2);
  c.u[1] = *(const uint4*)(rowp + kk + hi8 * 2 + 8);
  return c.v;
}

// B-fragment (32x16 bf16): lane l holds column (l&15); lanes<16: K=[kk..kk+15],
// lanes>=16: K=[kk+16..kk+31] (16 consecutive).  Two 16B loads.
__device__ __forceinline__ bf16x16 fragB(const bf16_t* colp, int kk, int hi16) {
  union { uint4 u[2]; bf16x16 v; } c;
  c.u[0] = *(const uint4*)(colp + kk + hi16);
  c.u[1] = *(const uint4*)(colp + kk + hi16 + 8);
  return c.v;
}

// --------------------------- weight prep ----------------------------------
__global__ void odelstm_prep(const float* __restrict__ Wih, const float* __restrict__ Whh,
                             const float* __restrict__ W1,  const float* __restrict__ W2,
                             const float* __restrict__ bih, const float* __restrict__ bhh,
                             bf16_t* wihB, bf16_t* whhB, bf16_t* w1B, bf16_t* w2B,
                             float* bsum) {
  const int NW0 = 4 * H_DIM * I_DIM;   // 524288
  const int NW1 = 4 * H_DIM * H_DIM;   // 1048576
  const int NW2 = H_DIM * H_DIM;       // 262144
  const int TOT = NW0 + NW1 + 2 * NW2 + 4 * H_DIM;
  const int stride = gridDim.x * blockDim.x;
  for (int i = blockIdx.x * blockDim.x + threadIdx.x; i < TOT; i += stride) {
    int j = i;
    if (j < NW0) { wihB[j] = (bf16_t)Wih[j]; continue; }
    j -= NW0;
    if (j < NW1) { whhB[j] = (bf16_t)Whh[j]; continue; }
    j -= NW1;
    if (j < NW2) { w1B[j] = (bf16_t)W1[j]; continue; }
    j -= NW2;
    if (j < NW2) { w2B[j] = (bf16_t)W2[j]; continue; }
    j -= NW2;
    bsum[j] = bih[j] + bhh[j];
  }
}

// --------------------------- fused main kernel ----------------------------
__global__ __launch_bounds__(256)
void odelstm_fused(const float* __restrict__ x,
                   const float* __restrict__ h0,
                   const float* __restrict__ c0,
                   const float* __restrict__ ts,
                   const bf16_t* __restrict__ WihB,
                   const bf16_t* __restrict__ WhhB,
                   const bf16_t* __restrict__ W1B,
                   const bf16_t* __restrict__ W2B,
                   const float* __restrict__ bsum,
                   const float* __restrict__ b1,
                   const float* __restrict__ b2,
                   float* __restrict__ out_ht,
                   float* __restrict__ out_ct) {
  __shared__ __align__(16) bf16_t xA [MT][XPAD];   // ~16.9 KB
  __shared__ __align__(16) bf16_t hP0[MT][HPAD];   // ~33.3 KB (h ping)
  __shared__ __align__(16) bf16_t hP1[MT][HPAD];   // ~33.3 KB (h pong)
  __shared__ __align__(16) float  cT [MT][H_DIM];  // 64 KB
  __shared__            float  dtS[MT];

  const int tid = threadIdx.x;
  const int b0  = blockIdx.x * MT;

  // -------- cooperative load: x,h0 -> bf16 LDS; c0 -> f32 LDS; dt --------
  for (int i = tid; i < MT * I_DIM; i += 256) {
    int r = i >> 8, c = i & (I_DIM - 1);
    xA[r][c] = (bf16_t)x[(size_t)(b0 + r) * I_DIM + c];
  }
  for (int i = tid; i < MT * H_DIM; i += 256) {
    int r = i >> 9, c = i & (H_DIM - 1);
    hP0[r][c] = (bf16_t)h0[(size_t)(b0 + r) * H_DIM + c];
    cT[r][c]  = c0[(size_t)(b0 + r) * H_DIM + c];
  }
  if (tid < MT) dtS[tid] = ts[(b0 + tid) * 2 + 1] - ts[(b0 + tid) * 2 + 0];
  __syncthreads();

  const int lane = tid & 31;
  const int wid  = tid >> 5;
  const int wm   = (wid & 1) * 16;     // M base (rows 0-15 / 16-31)
  const int wn   = (wid >> 1) * 128;   // N base within H (4 slices)
  const int l15  = lane & 15;
  const int hi8  = (lane >> 4) * 8;    // A-frag K selector
  const int hi16 = (lane >> 4) * 16;   // B-frag K selector
  const int rsel = (lane >> 4) * 8;    // C-frag row offset

  float dtv[8];
  #pragma unroll
  for (int r = 0; r < 8; ++r) dtv[r] = dtS[wm + r + rsel];

  const f32x8 zero8 = {0.f, 0.f, 0.f, 0.f, 0.f, 0.f, 0.f, 0.f};
  f32x8 newh[8];   // wave-private new_h f32 fragments (stage 2 output)

  // ---------------- two LSTM cell stages ----------------------------------
  #pragma unroll
  for (int stage = 0; stage < 2; ++stage) {
    const bf16_t* hsrc = (stage == 0) ? &hP0[0][0] : &hP1[0][0];
    bf16_t*       hdst = (stage == 0) ? &hP1[0][0] : &hP0[0][0];
    #pragma unroll
    for (int t = 0; t < 8; ++t) {
      const int ncol = wn + t * 16 + l15;          // output column (per lane)
      f32x8 acc[4];
      acc[0] = zero8; acc[1] = zero8; acc[2] = zero8; acc[3] = zero8;

      const bf16_t* arow_x = &xA[wm + l15][0];
      #pragma unroll 4
      for (int kk = 0; kk < I_DIM; kk += 32) {
        bf16x16 a = fragA(arow_x, kk, hi8);
        #pragma unroll
        for (int g = 0; g < 4; ++g) {
          bf16x16 b = fragB(WihB + (size_t)(g * H_DIM + ncol) * I_DIM, kk, hi16);
          acc[g] = wmma_bf16(a, b, acc[g]);
        }
      }
      const bf16_t* arow_h = hsrc + (wm + l15) * HPAD;
      #pragma unroll 4
      for (int kk = 0; kk < H_DIM; kk += 32) {
        bf16x16 a = fragA(arow_h, kk, hi8);
        #pragma unroll
        for (int g = 0; g < 4; ++g) {
          bf16x16 b = fragB(WhhB + (size_t)(g * H_DIM + ncol) * H_DIM, kk, hi16);
          acc[g] = wmma_bf16(a, b, acc[g]);
        }
      }
      const float bi = bsum[0 * H_DIM + ncol];
      const float bf_ = bsum[1 * H_DIM + ncol];
      const float bg = bsum[2 * H_DIM + ncol];
      const float bo = bsum[3 * H_DIM + ncol];
      #pragma unroll
      for (int r = 0; r < 8; ++r) {
        const int m = wm + r + rsel;
        const float iv = sigf(acc[0][r] + bi);
        const float fv = sigf(acc[1][r] + bf_);
        const float gv = fast_tanh(acc[2][r] + bg);
        const float ov = sigf(acc[3][r] + bo);
        const float cnew = fv * cT[m][wn + t * 16 + l15] + iv * gv;
        const float hnew = ov * fast_tanh(cnew);
        hdst[m * HPAD + wn + t * 16 + l15] = (bf16_t)hnew;
        if (stage == 0) {
          cT[m][wn + t * 16 + l15] = cnew;             // c1 for stage 2
        } else {
          newh[t][r] = hnew;                            // keep f32 new_h
          out_ct[(size_t)(b0 + m) * H_DIM + wn + t * 16 + l15] = cnew; // final c
        }
      }
    }
    __syncthreads();
  }

  // ---------------- RK4: f(h) = tanh(h@W1^T+b1)@W2^T + b2 -----------------
  f32x8 ksum[8];
  #pragma unroll
  for (int t = 0; t < 8; ++t) ksum[t] = zero8;

  for (int rk = 0; rk < 4; ++rk) {                 // uniform rolled loop
    // u = tanh(h_cur @ W1^T + b1): read hP0, write u -> hP1
    #pragma unroll
    for (int t = 0; t < 8; ++t) {
      const int ncol = wn + t * 16 + l15;
      f32x8 acc = zero8;
      const bf16_t* arow = &hP0[0][0] + (wm + l15) * HPAD;
      #pragma unroll 4
      for (int kk = 0; kk < H_DIM; kk += 32) {
        bf16x16 a = fragA(arow, kk, hi8);
        bf16x16 b = fragB(W1B + (size_t)ncol * H_DIM, kk, hi16);
        acc = wmma_bf16(a, b, acc);
      }
      const float bb = b1[ncol];
      #pragma unroll
      for (int r = 0; r < 8; ++r) {
        const int m = wm + r + rsel;
        hP1[m][wn + t * 16 + l15] = (bf16_t)fast_tanh(acc[r] + bb);
      }
    }
    __syncthreads();

    // k = u @ W2^T + b2: read hP1; ksum += w*k; next h_arg -> hP0
    const float kw = (rk == 1 || rk == 2) ? 2.0f : 1.0f;
    const float al = (rk < 2) ? 0.5f : 1.0f;
    #pragma unroll
    for (int t = 0; t < 8; ++t) {
      const int ncol = wn + t * 16 + l15;
      f32x8 acc = zero8;
      const bf16_t* arow = &hP1[0][0] + (wm + l15) * HPAD;
      #pragma unroll 4
      for (int kk = 0; kk < H_DIM; kk += 32) {
        bf16x16 a = fragA(arow, kk, hi8);
        bf16x16 b = fragB(W2B + (size_t)ncol * H_DIM, kk, hi16);
        acc = wmma_bf16(a, b, acc);
      }
      const float bb = b2[ncol];
      #pragma unroll
      for (int r = 0; r < 8; ++r) {
        const float kv = acc[r] + bb;
        ksum[t][r] += kw * kv;
        if (rk < 3) {
          const int m = wm + r + rsel;
          hP0[m][wn + t * 16 + l15] = (bf16_t)(newh[t][r] + al * dtv[r] * kv);
        }
      }
    }
    __syncthreads();
  }

  // ---------------- final combine: ht = new_h + dt/6 * ksum ---------------
  #pragma unroll
  for (int t = 0; t < 8; ++t) {
    #pragma unroll
    for (int r = 0; r < 8; ++r) {
      const int m = wm + r + rsel;
      out_ht[(size_t)(b0 + m) * H_DIM + wn + t * 16 + l15] =
          newh[t][r] + (dtv[r] * (1.0f / 6.0f)) * ksum[t][r];
    }
  }
}

// --------------------------- host launch ----------------------------------
extern "C" void kernel_launch(void* const* d_in, const int* in_sizes, int n_in,
                              void* d_out, int out_size, void* d_ws, size_t ws_size,
                              hipStream_t stream) {
  const float* x   = (const float*)d_in[0];
  const float* h0  = (const float*)d_in[1];
  const float* c0  = (const float*)d_in[2];
  const float* ts  = (const float*)d_in[3];
  const float* Wih = (const float*)d_in[4];
  const float* Whh = (const float*)d_in[5];
  const float* bih = (const float*)d_in[6];
  const float* bhh = (const float*)d_in[7];
  const float* W1  = (const float*)d_in[8];
  const float* b1  = (const float*)d_in[9];
  const float* W2  = (const float*)d_in[10];
  const float* b2  = (const float*)d_in[11];

  // workspace layout (bf16 weights + fused bias), ~4.2 MB
  char*   ws   = (char*)d_ws;
  bf16_t* wihB = (bf16_t*)(ws);                     // 2048*256 bf16
  bf16_t* whhB = (bf16_t*)(ws + 1048576);           // 2048*512 bf16
  bf16_t* w1B  = (bf16_t*)(ws + 3145728);           // 512*512 bf16
  bf16_t* w2B  = (bf16_t*)(ws + 3670016);           // 512*512 bf16
  float*  bsum = (float*)(ws + 4194304);            // 2048 f32

  odelstm_prep<<<1024, 256, 0, stream>>>(Wih, Whh, W1, W2, bih, bhh,
                                         wihB, whhB, w1B, w2B, bsum);

  float* out_ht = (float*)d_out;
  float* out_ct = out_ht + (size_t)B_TOT * H_DIM;
  odelstm_fused<<<B_TOT / MT, 256, 0, stream>>>(x, h0, c0, ts,
                                                wihB, whhB, w1B, w2B,
                                                bsum, b1, b2, out_ht, out_ct);
}
